// Gate_51771535786338
// MI455X (gfx1250) — matrix operand order repaired and checked by reading, not compiled, not run on hardware
//
#include <hip/hip_runtime.h>
#include <hip/hip_bf16.h>

typedef __attribute__((ext_vector_type(16))) __bf16 v16bf;
typedef __attribute__((ext_vector_type(8)))  float  v8f;

#define D_EMB     4096
#define N_EXP     64
#define TOPK      4
#define N_GRP     8
#define EPG       8
#define LIM_GRP   4

#define WAVES         4
#define MT            1                       // 16 tokens per wave
#define NT            4                       // 4 N-tiles = 64 experts
#define TOK_PER_WAVE  16
#define TOK_PER_BLOCK (WAVES * TOK_PER_WAVE)  // 64 tokens per block
#define KSTEP         32
#define NKS           (D_EMB / KSTEP)         // 128 K-steps
#define LSTRIDE       (N_EXP + 1)             // padded logits row (floats)
#define WSTRIDE       40                      // padded W-slice row (bf16): 80B, 16B-aligned
#define WBF_ELEMS     (N_EXP * D_EMB)         // 262144 halves = 512 KB

// ---- prep: zero counts + convert W fp32 -> bf16 into scratch ----
__global__ void prep_kernel(const float* __restrict__ W, __bf16* __restrict__ wbf,
                            int* __restrict__ counts, int do_conv) {
    if (blockIdx.x == 0 && threadIdx.x < N_EXP) counts[threadIdx.x] = 0;
    if (do_conv) {
        const int stride = gridDim.x * blockDim.x;
        for (int i = blockIdx.x * blockDim.x + threadIdx.x; i < WBF_ELEMS; i += stride)
            wbf[i] = (__bf16)W[i];
    }
}

__device__ __forceinline__ unsigned lds_off_u32(const void* p) {
    // low 32 bits of the flat LDS-aperture address == LDS byte address
    return (unsigned)(uintptr_t)p;
}

__device__ __forceinline__ void async_ld_b128(unsigned lds_off, unsigned gl_off,
                                              const void* sbase) {
    asm volatile("global_load_async_to_lds_b128 %0, %1, %2"
                 :
                 : "v"(lds_off), "v"(gl_off), "s"(sbase)
                 : "memory");
}

__device__ __forceinline__ void cvt4(v16bf& v, int base, float4 f) {
    v[base + 0] = (__bf16)f.x;
    v[base + 1] = (__bf16)f.y;
    v[base + 2] = (__bf16)f.z;
    v[base + 3] = (__bf16)f.w;
}

template <bool PRECONV>
__global__ void __launch_bounds__(WAVES * 32)
moe_gate_kernel(const float* __restrict__ x, const float* __restrict__ W,
                const __bf16* __restrict__ wbf,
                float* __restrict__ out_w, int* __restrict__ out_idx,
                int* __restrict__ counts, int n_tokens)
{
    __shared__ float  lds_logits[WAVES][TOK_PER_WAVE][LSTRIDE];     // 16.6 KB
    __shared__ __bf16 lds_w[2][N_EXP * WSTRIDE];                    // 2 x 5 KB

    const int tid  = threadIdx.x;
    const int wave = tid >> 5;
    const int lane = tid & 31;
    const int half = lane >> 4;   // 0: lanes 0-15, 1: lanes 16-31
    const int l16  = lane & 15;

    const int tok0 = blockIdx.x * TOK_PER_BLOCK + wave * TOK_PER_WAVE;

    v8f acc[NT];
#pragma unroll
    for (int nt = 0; nt < NT; ++nt)
        acc[nt] = (v8f){0.f, 0.f, 0.f, 0.f, 0.f, 0.f, 0.f, 0.f};

    // A-fragment row (ISA 16-bit A layout: both lane halves cover M=0..15)
    int r = tok0 + l16;
    if (r > n_tokens - 1) r = n_tokens - 1;          // clamp, EXEC stays uniform
    const float* xrow = x + (size_t)r * D_EMB;
    const int aoff0 = half * 16;  // lane<16: K{0..15}; lane>=16: K{16..31}

    // async W staging: 4 KB slice = 256 x 16B chunks, 128 threads x 2 chunks
    const int c0 = tid * 2;                           // chunk ids c0, c0+1
    const int e0 = c0 >> 2, s0 = c0 & 3;              // expert row, 16B sub-chunk
    const int e1 = (c0 + 1) >> 2, s1 = (c0 + 1) & 3;

    if (PRECONV) {   // prologue: stage slice 0 into buffer 0
        async_ld_b128(lds_off_u32(&lds_w[0][e0 * WSTRIDE + s0 * 8]),
                      (unsigned)(e0 * D_EMB * 2 + s0 * 16), wbf);
        async_ld_b128(lds_off_u32(&lds_w[0][e1 * WSTRIDE + s1 * 8]),
                      (unsigned)(e1 * D_EMB * 2 + s1 * 16), wbf);
    }

    for (int ks = 0; ks < NKS; ++ks) {
        const int kb  = ks * KSTEP;
        const int buf = PRECONV ? (ks & 1) : 0;

        if (PRECONV) {
            if (ks + 1 < NKS) {   // issue next slice into other buffer, then wait current
                const unsigned gb = (unsigned)((ks + 1) * KSTEP * 2);
                async_ld_b128(lds_off_u32(&lds_w[buf ^ 1][e0 * WSTRIDE + s0 * 8]),
                              (unsigned)(e0 * D_EMB * 2) + gb + s0 * 16, wbf);
                async_ld_b128(lds_off_u32(&lds_w[buf ^ 1][e1 * WSTRIDE + s1 * 8]),
                              (unsigned)(e1 * D_EMB * 2) + gb + s1 * 16, wbf);
                asm volatile("s_wait_asynccnt 2" ::: "memory");
            } else {
                asm volatile("s_wait_asynccnt 0" ::: "memory");
            }
            __syncthreads();      // slice ks visible to all waves
        } else {
            __syncthreads();      // previous B reads done before restaging
            // synchronous staging with on-the-fly fp32->bf16 (fallback path)
            const int e  = tid >> 1;
            const int kh = (tid & 1) * 16;
            const float4* src = (const float4*)(W + (size_t)e * D_EMB + kb + kh);
            float4 f0 = src[0], f1 = src[1], f2 = src[2], f3 = src[3];
            union { __bf16 h[16]; uint4 q[2]; } u;
            u.h[0]  = (__bf16)f0.x; u.h[1]  = (__bf16)f0.y; u.h[2]  = (__bf16)f0.z; u.h[3]  = (__bf16)f0.w;
            u.h[4]  = (__bf16)f1.x; u.h[5]  = (__bf16)f1.y; u.h[6]  = (__bf16)f1.z; u.h[7]  = (__bf16)f1.w;
            u.h[8]  = (__bf16)f2.x; u.h[9]  = (__bf16)f2.y; u.h[10] = (__bf16)f2.z; u.h[11] = (__bf16)f2.w;
            u.h[12] = (__bf16)f3.x; u.h[13] = (__bf16)f3.y; u.h[14] = (__bf16)f3.z; u.h[15] = (__bf16)f3.w;
            uint4* dst = (uint4*)&lds_w[0][e * WSTRIDE + kh];
            dst[0] = u.q[0];
            dst[1] = u.q[1];
            __syncthreads();
        }

        // A fragment: 4x float4, convert to bf16
        v16bf afrag;
        {
            const float* p = xrow + kb + aoff0;
            cvt4(afrag, 0,  ((const float4*)p)[0]);
            cvt4(afrag, 4,  ((const float4*)(p + 4))[0]);
            cvt4(afrag, 8,  ((const float4*)(p + 8))[0]);
            cvt4(afrag, 12, ((const float4*)(p + 12))[0]);
            if (kb + KSTEP < D_EMB)
                __builtin_prefetch(p + KSTEP, 0, 0);   // next A slice -> global_prefetch_b8
        }

        // B fragments from LDS: lane holds column n = l16, K-half per lane group
        v16bf bfrag[NT];
#pragma unroll
        for (int nt = 0; nt < NT; ++nt) {
            const int e = nt * 16 + l16;
            union { v16bf v; uint4 q[2]; } bu;
            const uint4* p = (const uint4*)&lds_w[buf][e * WSTRIDE + half * 16];
            bu.q[0] = p[0];
            bu.q[1] = p[1];
            bfrag[nt] = bu.v;
        }

#pragma unroll
        for (int nt = 0; nt < NT; ++nt)
            acc[nt] = __builtin_amdgcn_wmma_f32_16x16x32_bf16(
                false, afrag, false, bfrag[nt], (short)0, acc[nt], false, false);

        __syncthreads();          // all B reads done before buffer is rewritten
    }

    // ---- dump logits (C layout: VGPR i -> M=i lanes 0-15 / M=i+8 lanes 16-31) ----
#pragma unroll
    for (int nt = 0; nt < NT; ++nt) {
        const int e = nt * 16 + l16;
#pragma unroll
        for (int i = 0; i < 8; ++i)
            lds_logits[wave][i + half * 8][e] = acc[nt][i];
    }
    __syncthreads();

    // ---- routing: lanes 0-15 each own one token ----
    const int token = tok0 + lane;
    if (lane < TOK_PER_WAVE && token < n_tokens) {
        float* row = lds_logits[wave][lane];

        float mx = row[0];
#pragma unroll
        for (int e = 1; e < N_EXP; ++e) mx = fmaxf(mx, row[e]);

        float sum = 0.f;
#pragma unroll
        for (int e = 0; e < N_EXP; ++e) {
            float p = __expf(row[e] - mx);
            row[e] = p;                       // unnormalized score
            sum += p;
        }
        const float inv = 1.0f / sum;         // softmax norm; ROUTE_SCALE = 1

        float gs[N_GRP];
#pragma unroll
        for (int g = 0; g < N_GRP; ++g) {
            float m = row[g * EPG];
#pragma unroll
            for (int j = 1; j < EPG; ++j) m = fmaxf(m, row[g * EPG + j]);
            gs[g] = m;
        }

        bool sel[N_GRP];
#pragma unroll
        for (int g = 0; g < N_GRP; ++g) sel[g] = false;
#pragma unroll
        for (int rr = 0; rr < LIM_GRP; ++rr) {
            int bg = -1; float bv = -1.f;
#pragma unroll
            for (int g = 0; g < N_GRP; ++g)
                if (!sel[g] && gs[g] > bv) { bv = gs[g]; bg = g; }
            sel[bg] = true;
        }

        float tv[TOPK] = {-1.f, -1.f, -1.f, -1.f};
        int   ti[TOPK] = {0, 0, 0, 0};
#pragma unroll
        for (int e = 0; e < N_EXP; ++e) {
            if (!sel[e >> 3]) continue;
            const float v = row[e];
            if (v > tv[0]) {
                tv[3]=tv[2]; ti[3]=ti[2]; tv[2]=tv[1]; ti[2]=ti[1];
                tv[1]=tv[0]; ti[1]=ti[0]; tv[0]=v; ti[0]=e;
            } else if (v > tv[1]) {
                tv[3]=tv[2]; ti[3]=ti[2]; tv[2]=tv[1]; ti[2]=ti[1];
                tv[1]=v; ti[1]=e;
            } else if (v > tv[2]) {
                tv[3]=tv[2]; ti[3]=ti[2]; tv[2]=v; ti[2]=e;
            } else if (v > tv[3]) {
                tv[3]=v; ti[3]=e;
            }
        }

#pragma unroll
        for (int j = 0; j < TOPK; ++j) {
            out_w[(size_t)token * TOPK + j]   = tv[j] * inv;
            out_idx[(size_t)token * TOPK + j] = ti[j];
            atomicAdd(&counts[ti[j]], 1);
        }
    }
}

extern "C" void kernel_launch(void* const* d_in, const int* in_sizes, int n_in,
                              void* d_out, int out_size, void* d_ws, size_t ws_size,
                              hipStream_t stream) {
    const float* x = (const float*)d_in[0];
    const float* W = (const float*)d_in[1];
    const int N = in_sizes[0] / D_EMB;   // 16384

    float* out_w   = (float*)d_out;
    int*   out_idx = (int*)((float*)d_out + (size_t)N * TOPK);
    int*   counts  = (int*)((float*)d_out + (size_t)2 * N * TOPK);

    const bool preconv = (d_ws != nullptr) && (ws_size >= (size_t)WBF_ELEMS * sizeof(__bf16));
    __bf16* wbf = (__bf16*)d_ws;

    prep_kernel<<<128, 256, 0, stream>>>(W, wbf, counts, preconv ? 1 : 0);

    dim3 grid((N + TOK_PER_BLOCK - 1) / TOK_PER_BLOCK);
    if (preconv)
        moe_gate_kernel<true ><<<grid, WAVES * 32, 0, stream>>>(x, W, wbf, out_w, out_idx, counts, N);
    else
        moe_gate_kernel<false><<<grid, WAVES * 32, 0, stream>>>(x, W, wbf, out_w, out_idx, counts, N);
}
